// Attention_1958505087484
// MI455X (gfx1250) — matrix operand order repaired and checked by reading, not compiled, not run on hardware
//
#include <hip/hip_runtime.h>
#include <hip/hip_bf16.h>

// ---------- problem constants ----------
#define BB   32
#define SS   2048
#define HH   1024
#define OUTK 1024

// ---------- WMMA vector types ----------
typedef __attribute__((ext_vector_type(16))) __bf16 v16bf;
typedef __attribute__((ext_vector_type(8)))  __bf16 v8bf;
typedef __attribute__((ext_vector_type(8)))  float  v8f;

__device__ __forceinline__ unsigned short f2bf(float f) {
    unsigned int x = __float_as_uint(f);
    unsigned int r = x + 0x7FFFu + ((x >> 16) & 1u);   // round-to-nearest-even
    return (unsigned short)(r >> 16);
}

__device__ __forceinline__ v16bf combine16(v8bf lo, v8bf hi) {
    return __builtin_shufflevector(lo, hi, 0,1,2,3,4,5,6,7,8,9,10,11,12,13,14,15);
}

// Branch-free tanh: 2 trans ops (exp2, rcp) + ~4 VALU, no EXEC divergence.
// tanh(|x|) = 1 - 2/(e^{2|x|}+1); e^{2|x|} -> +inf saturates cleanly to 1.
__device__ __forceinline__ float fast_tanh(float x) {
    float ax = __builtin_fabsf(x);
    float e  = __builtin_amdgcn_exp2f(ax * 2.8853900817779268f); // 2*log2(e)
    float t  = 1.0f - 2.0f * __builtin_amdgcn_rcpf(e + 1.0f);
    return __builtin_copysignf(t, x);
}

// ============================================================
// Kernel P: pack W_a (HxH f32) into per-lane WMMA-B bf16 layout.
// Tile (kt,nt) is 32x16 (KxN).  B-frag per lane L: col n = nt*16 + (L&15),
// 16 contiguous k values k = kt*32 + 16*(L>>4) + j  (j=0..15).
// Packed order: tile = nt*32 + kt, element = tile*512 + L*16 + j  (u16)
// ============================================================
__global__ void pack_wa_kernel(const float* __restrict__ W_a,
                               unsigned short* __restrict__ pB) {
    int idx  = blockIdx.x * 256 + threadIdx.x;       // 0 .. 1M-1
    int j    = idx & 15;
    int lane = (idx >> 4) & 31;
    int tile = idx >> 9;
    int kt   = tile & 31;
    int nt   = tile >> 5;
    int k = kt * 32 + 16 * (lane >> 4) + j;
    int n = nt * 16 + (lane & 15);
    pB[idx] = f2bf(W_a[(size_t)k * HH + n]);
}

// ============================================================
// Kernel 1: tanh_hp = tanh(h_t @ W_p)   and   wt = h_t @ U_a
// grid (4, B, 2), block 256
// ============================================================
__global__ void ht_gemm_kernel(const float* __restrict__ inputs,
                               const float* __restrict__ W_p,
                               const float* __restrict__ U_a,
                               float* __restrict__ tanh_hp,
                               float* __restrict__ wt) {
    int k = blockIdx.x * 256 + threadIdx.x;
    int b = blockIdx.y;
    const float* W = (blockIdx.z == 0) ? W_p : U_a;
    const float* ht = inputs + ((size_t)b * SS + (SS - 1)) * HH;
    float acc = 0.f;
    for (int h = 0; h < HH; ++h)
        acc = fmaf(ht[h], W[(size_t)h * HH + k], acc);
    if (blockIdx.z == 0) tanh_hp[b * HH + k] = fast_tanh(acc);
    else                 wt[b * HH + k] = acc;
}

// ============================================================
// Kernel 1b: aligned_position[b] = sigmoid(tanh_hp[b] . v_p) * S
// ============================================================
__global__ void align_pos_kernel(const float* __restrict__ tanh_hp,
                                 const float* __restrict__ v_p,
                                 float* __restrict__ apos) {
    __shared__ float red[256];
    int b = blockIdx.x, tid = threadIdx.x;
    float acc = 0.f;
    for (int h = tid; h < HH; h += 256)
        acc = fmaf(tanh_hp[b * HH + h], v_p[h], acc);
    red[tid] = acc; __syncthreads();
    for (int o = 128; o > 0; o >>= 1) {
        if (tid < o) red[tid] += red[tid + o];
        __syncthreads();
    }
    if (tid == 0) apos[b] = (1.0f / (1.0f + expf(-red[0]))) * (float)SS;
}

// ============================================================
// Kernel 2 (core): fused   score[b,s] = sum_n tanh( (inputs@W_a)[b,s,n]
//                                                   + wt[b,n] ) * v_a[n]
// M = B*S rows, N = K = 1024.  bf16 WMMA 16x16x32, f32 accumulate.
// block = 128 threads (4 waves), 64-row strip per block, grid = 1024.
// Two n-tiles per wave pass: two INDEPENDENT accumulator chains (breaks
// the WMMA D->C serial dependence) and 2x reuse of each A fragment
// (ds_load : wmma drops from 2:1 to 1:1).
// Dynamic LDS: A strip bf16 (stride-padded) + wt row + v_a = 140288 B.
// ============================================================
#define LDS_STRIDE 1032   // 1024 + 8 bf16 pad

__global__ __launch_bounds__(128)
void score_kernel(const float* __restrict__ inputs,
                  const unsigned short* __restrict__ pB,
                  const float* __restrict__ wt,
                  const float* __restrict__ v_a,
                  float* __restrict__ score) {
    extern __shared__ __align__(16) unsigned char smemRaw[];
    unsigned short* ldsA = (unsigned short*)smemRaw;              // 64*1032 u16
    float* ldsWT = (float*)(smemRaw + 64 * LDS_STRIDE * 2);       // 1024 f32
    float* ldsVA = ldsWT + HH;                                    // 1024 f32

    const int tid  = threadIdx.x;
    const int row0 = blockIdx.x * 64;            // flattened b*S + s
    const int b    = row0 >> 11;                 // S = 2048

    // stage wt[b,:] and v_a
    for (int i = tid; i < HH; i += 128) {
        ldsWT[i] = wt[b * HH + i];
        ldsVA[i] = v_a[i];
    }
    // stage A strip: 64 rows x 1024 f32 -> bf16 LDS (pairs packed as u32)
    for (int idx = tid; idx < 64 * (HH / 4); idx += 128) {
        int r  = idx >> 8;           // / 256
        int c4 = idx & 255;
        const float4 f = ((const float4*)(inputs + ((size_t)(row0 + r)) * HH))[c4];
        unsigned int lo = (unsigned int)f2bf(f.x) | ((unsigned int)f2bf(f.y) << 16);
        unsigned int hi = (unsigned int)f2bf(f.z) | ((unsigned int)f2bf(f.w) << 16);
        uint2 pk; pk.x = lo; pk.y = hi;
        *(uint2*)(&ldsA[r * LDS_STRIDE + c4 * 4]) = pk;
    }
    __syncthreads();

    const int wave = tid >> 5;
    const int lane = tid & 31;
    const int half = lane >> 4;
    const int lr   = lane & 15;
    const int wrow = wave * 16;

    const unsigned short* aRow = ldsA + (size_t)(wrow + lr) * LDS_STRIDE;

    float rs[8];
#pragma unroll
    for (int v = 0; v < 8; ++v) rs[v] = 0.f;

    for (int nt = 0; nt < 64; nt += 2) {
        v8f c0 = {0.f, 0.f, 0.f, 0.f, 0.f, 0.f, 0.f, 0.f};
        v8f c1 = {0.f, 0.f, 0.f, 0.f, 0.f, 0.f, 0.f, 0.f};
        const unsigned short* bp0 = pB + (size_t)nt * 32 * 512 + lane * 16;
        const unsigned short* bp1 = bp0 + 32 * 512;
#pragma unroll 4
        for (int kt = 0; kt < 32; ++kt) {
            const unsigned short* ap = aRow + kt * 32 + 16 * half;
            v8bf alo = *(const v8bf*)(ap);
            v8bf ahi = *(const v8bf*)(ap + 8);
            v16bf a  = combine16(alo, ahi);

            v8bf b0lo = *(const v8bf*)(bp0);
            v8bf b0hi = *(const v8bf*)(bp0 + 8);
            v8bf b1lo = *(const v8bf*)(bp1);
            v8bf b1hi = *(const v8bf*)(bp1 + 8);
            v16bf bm0 = combine16(b0lo, b0hi);
            v16bf bm1 = combine16(b1lo, b1hi);

            c0 = __builtin_amdgcn_wmma_f32_16x16x32_bf16(
                     false, a, false, bm0, (short)0, c0, false, false);
            c1 = __builtin_amdgcn_wmma_f32_16x16x32_bf16(
                     false, a, false, bm1, (short)0, c1, false, false);
            bp0 += 512;
            bp1 += 512;
        }
        // fused epilogue: tanh + dot with v_a, per-row partials (branch-free)
        const float wtv0 = ldsWT[nt * 16 + lr];
        const float vav0 = ldsVA[nt * 16 + lr];
        const float wtv1 = ldsWT[(nt + 1) * 16 + lr];
        const float vav1 = ldsVA[(nt + 1) * 16 + lr];
#pragma unroll
        for (int v = 0; v < 8; ++v) {
            rs[v] += fast_tanh(c0[v] + wtv0) * vav0;
            rs[v] += fast_tanh(c1[v] + wtv1) * vav1;
        }
    }
    // reduce across the 16 lanes of each half
#pragma unroll
    for (int m = 1; m < 16; m <<= 1)
#pragma unroll
        for (int v = 0; v < 8; ++v)
            rs[v] += __shfl_xor(rs[v], m, 32);

    if (lr == 0) {
#pragma unroll
        for (int v = 0; v < 8; ++v)
            score[row0 + wrow + half * 8 + v] = rs[v];
    }
}

// ============================================================
// Kernel 3: attention = softmax(score) * gaussian(pos, aligned_pos)
// one block per batch
// ============================================================
__global__ void softmax_gauss_kernel(const float* __restrict__ score,
                                     const float* __restrict__ apos,
                                     float* __restrict__ aw) {
    __shared__ float red[256];
    int b = blockIdx.x, tid = threadIdx.x;
    const float* sc = score + (size_t)b * SS;

    float m = -1e30f;
    for (int s = tid; s < SS; s += 256) m = fmaxf(m, sc[s]);
    red[tid] = m; __syncthreads();
    for (int o = 128; o > 0; o >>= 1) {
        if (tid < o) red[tid] = fmaxf(red[tid], red[tid + o]);
        __syncthreads();
    }
    float mx = red[0]; __syncthreads();

    float sum = 0.f;
    for (int s = tid; s < SS; s += 256) sum += expf(sc[s] - mx);
    red[tid] = sum; __syncthreads();
    for (int o = 128; o > 0; o >>= 1) {
        if (tid < o) red[tid] += red[tid + o];
        __syncthreads();
    }
    float inv_tot = 1.0f / red[0];
    float ap = apos[b];
    const float inv_denom = 1.0f / 524288.0f;   // 2*(S/4)^2 = 2*512^2
    for (int s = tid; s < SS; s += 256) {
        float d = (float)s - ap;
        aw[(size_t)b * SS + s] =
            expf(sc[s] - mx) * inv_tot * expf(-d * d * inv_denom);
    }
}

// ============================================================
// Kernel 4: context partials (deterministic two-stage, no fp atomics)
// grid (16 s-chunks, B), block 256; each thread owns 4 h values
// ============================================================
__global__ void ctx_partial_kernel(const float* __restrict__ inputs,
                                   const float* __restrict__ aw,
                                   float* __restrict__ part) {
    int chunk = blockIdx.x, b = blockIdx.y, tid = threadIdx.x;
    float acc0 = 0.f, acc1 = 0.f, acc2 = 0.f, acc3 = 0.f;
    int s0 = chunk * (SS / 16);
    for (int s = s0; s < s0 + SS / 16; ++s) {
        float w = aw[(size_t)b * SS + s];
        const float* row = inputs + ((size_t)b * SS + s) * HH;
        acc0 = fmaf(row[tid +   0], w, acc0);
        acc1 = fmaf(row[tid + 256], w, acc1);
        acc2 = fmaf(row[tid + 512], w, acc2);
        acc3 = fmaf(row[tid + 768], w, acc3);
    }
    float* dst = part + ((size_t)chunk * BB + b) * HH;
    dst[tid +   0] = acc0; dst[tid + 256] = acc1;
    dst[tid + 512] = acc2; dst[tid + 768] = acc3;
}

__global__ void ctx_reduce_kernel(const float* __restrict__ part,
                                  float* __restrict__ ctx) {
    int g = blockIdx.x * 256 + threadIdx.x;      // 0 .. B*H-1
    int b = g >> 10, h = g & 1023;
    float acc = 0.f;
    for (int c = 0; c < 16; ++c)
        acc += part[((size_t)c * BB + b) * HH + h];
    ctx[g] = acc;
}

// ============================================================
// Kernel 5: out = tanh([context, h_t] @ W_att)   (32 x 2048 @ 2048 x 1024)
// ============================================================
__global__ void out_gemm_kernel(const float* __restrict__ ctx,
                                const float* __restrict__ inputs,
                                const float* __restrict__ W_att,
                                float* __restrict__ out) {
    int k = blockIdx.x * 256 + threadIdx.x;
    int b = blockIdx.y;
    const float* ht = inputs + ((size_t)b * SS + (SS - 1)) * HH;
    float acc = 0.f;
    for (int h = 0; h < HH; ++h)
        acc = fmaf(ctx[b * HH + h], W_att[(size_t)h * OUTK + k], acc);
    for (int h = 0; h < HH; ++h)
        acc = fmaf(ht[h], W_att[(size_t)(HH + h) * OUTK + k], acc);
    out[(size_t)b * OUTK + k] = fast_tanh(acc);
}

// ============================================================
// host launcher
// ============================================================
extern "C" void kernel_launch(void* const* d_in, const int* in_sizes, int n_in,
                              void* d_out, int out_size, void* d_ws, size_t ws_size,
                              hipStream_t stream) {
    (void)in_sizes; (void)n_in; (void)out_size; (void)ws_size;
    const float* inputs = (const float*)d_in[0];
    const float* W_p    = (const float*)d_in[1];
    const float* v_p    = (const float*)d_in[2];
    const float* W_a    = (const float*)d_in[3];
    const float* U_a    = (const float*)d_in[4];
    const float* v_a    = (const float*)d_in[5];
    const float* W_att  = (const float*)d_in[6];
    float* out = (float*)d_out;

    float* ws = (float*)d_ws;
    float* f_wt      = ws;                 // 32768
    float* f_thp     = ws + 32768;         // 32768
    float* f_apos    = ws + 65536;         // 64 (padded)
    float* f_score   = ws + 65600;         // 65536
    float* f_aw      = ws + 131136;        // 65536
    float* f_ctxpart = ws + 196672;        // 16*32768
    float* f_ctx     = ws + 720960;        // 32768
    unsigned short* pB = (unsigned short*)(ws + 753728);  // 1M bf16 (2 MB)

    pack_wa_kernel<<<4096, 256, 0, stream>>>(W_a, pB);
    ht_gemm_kernel<<<dim3(4, BB, 2), 256, 0, stream>>>(inputs, W_p, U_a, f_thp, f_wt);
    align_pos_kernel<<<BB, 256, 0, stream>>>(f_thp, v_p, f_apos);

    size_t smem = (size_t)64 * LDS_STRIDE * 2 + 2 * HH * sizeof(float); // 140288 B
    score_kernel<<<(BB * SS) / 64, 128, smem, stream>>>(inputs, pB, f_wt, v_a, f_score);

    softmax_gauss_kernel<<<BB, 256, 0, stream>>>(f_score, f_apos, f_aw);
    ctx_partial_kernel<<<dim3(16, BB), 256, 0, stream>>>(inputs, f_aw, f_ctxpart);
    ctx_reduce_kernel<<<(BB * HH) / 256, 256, 0, stream>>>(f_ctxpart, f_ctx);
    out_gemm_kernel<<<dim3(OUTK / 256, BB), 256, 0, stream>>>(f_ctx, inputs, W_att, out);
}
